// MutualForward_39616778338634
// MI455X (gfx1250) — matrix operand order repaired and checked by reading, not compiled, not run on hardware
//
#include <hip/hip_runtime.h>

#define NB      256      // graphs
#define NPGc    128      // nodes per graph
#define NNODE   32768    // NB*NPGc
#define HDIM    128
#define NLYR    3
#define L1c     4
#define NHEADc  4
#define HDHEAD  32
#define NEDGE   524288
#define ETOT    (NEDGE + NNODE)   // + self loops
#define EPSc    1e-5f

typedef _Float16 half_t;
typedef __attribute__((ext_vector_type(16))) _Float16 v16h;
typedef __attribute__((ext_vector_type(8)))  _Float16 v8h;
typedef __attribute__((ext_vector_type(8)))  float    v8f;

// compile-time for-loop so WMMA modifier bits are integer constant expressions
template <int I> struct IC { static constexpr int v = I; };
template <int I, int N, typename F>
__device__ __forceinline__ void sfor(F&& f) {
  if constexpr (I < N) { f(IC<I>{}); sfor<I + 1, N>(f); }
}

static __device__ __forceinline__ v16h cat8(v8h lo, v8h hi) {
  return __builtin_shufflevector(lo, hi, 0,1,2,3,4,5,6,7,8,9,10,11,12,13,14,15);
}

// A-matrix fragment load (contiguous K order, matching B):
// lane<16 : row M=lane,    K = {0..15}  of the 32-chunk
// lane>=16: row M=lane-16, K = {16..31}
static __device__ __forceinline__ v16h load_a_frag(const half_t* rowptr, int lane) {
  int ko = (lane < 16) ? 0 : 16;
  v8h lo = *(const v8h*)(rowptr + ko);
  v8h hi = *(const v8h*)(rowptr + 8 + ko);
  return cat8(lo, hi);
}

// ---------------------------------------------------------------------------
// float -> f16 cast
__global__ void k_cast(const float* __restrict__ a, half_t* __restrict__ o, int n) {
  int i = blockIdx.x * blockDim.x + threadIdx.x;
  if (i < n) o[i] = (half_t)a[i];
}

__global__ void k_zero(float* __restrict__ a, int n) {
  int i = blockIdx.x * blockDim.x + threadIdx.x;
  if (i < n) a[i] = 0.f;
}

// ---------------------------------------------------------------------------
// C[M,128] = A[M,128](f16) @ W.T + bias ; W is [128 out,128 in] f16 row-major.
// Block: 256 threads = 8 waves, each wave computes 16 rows x 128 cols.
// B fragments are software-pipelined from LDS; WMMAs in a K-chunk share A and
// carry the reuse-A operand-cache hint (VOP3P RA bit).
__global__ __launch_bounds__(256)
void k_gemm128(const half_t* __restrict__ A, const half_t* __restrict__ W,
               const float* __restrict__ bias, float* __restrict__ outF32,
               half_t* __restrict__ outF16, int act) {
  __shared__ half_t Wl[HDIM * HDIM];
  int tid = threadIdx.x;
  { // stage W into LDS (32KB), shared by all 8 waves
    const v16h* src = (const v16h*)W;
    v16h* dst = (v16h*)Wl;
    for (int i = tid; i < (HDIM * HDIM) / 16; i += 256) dst[i] = src[i];
  }
  __syncthreads();

  int wave = tid >> 5, lane = tid & 31;
  int lrow = lane & 15;
  int khalf = (lane < 16) ? 0 : 16;   // B-operand K half
  int rowbase = blockIdx.x * 128 + wave * 16;

  const half_t* arow = A + (size_t)(rowbase + lrow) * HDIM;
  __builtin_prefetch(arow, 0, 1);   // global_prefetch_b8

  v16h a[4];
#pragma unroll
  for (int kc = 0; kc < 4; kc++) a[kc] = load_a_frag(arow + kc * 32, lane);

  v8f acc[8];
#pragma unroll
  for (int j = 0; j < 8; j++) {
    float bv = bias ? bias[j * 16 + lrow] : 0.f;
    v8f t = {bv, bv, bv, bv, bv, bv, bv, bv};
    acc[j] = t;
  }

  // B fragment for col-tile j, K-chunk kc: lane<16 -> col n=j*16+lane, K 0..15
  // contiguous => row (j*16+lrow) of W.
  v16h bcur = *(const v16h*)(Wl + (size_t)lrow * HDIM + khalf);
  sfor<0, 4>([&](auto KC) {
    constexpr int kc = decltype(KC)::v;
    sfor<0, 8>([&](auto J) {
      constexpr int j = decltype(J)::v;
      v16h bnext = bcur;
      if constexpr (!(kc == 3 && j == 7)) {
        constexpr int jn = (j + 1) & 7;
        constexpr int kn = (j == 7) ? (kc + 1) : kc;
        bnext = *(const v16h*)(Wl + (jn * 16 + lrow) * HDIM + kn * 32 + khalf);
      }
      // reuse_a: next WMMA consumes the same A fragment for j<7
      acc[j] = __builtin_amdgcn_wmma_f32_16x16x32_f16(
          false, a[kc], false, bcur, (short)0, acc[j], (j < 7), false);
      bcur = bnext;
    });
  });

  int r0 = (lane < 16) ? 0 : 8;
#pragma unroll
  for (int j = 0; j < 8; j++) {
    int col = j * 16 + lrow;
#pragma unroll
    for (int r = 0; r < 8; r++) {
      float v = acc[j][r];
      if (act == 1) v = v > 0.f ? v : 0.f;
      size_t idx = (size_t)(rowbase + r0 + r) * HDIM + col;
      if (outF32) outF32[idx] = v;
      if (outF16) outF16[idx] = (half_t)v;
    }
  }
}

// ---------------------------------------------------------------------------
// Cross attention per (batch, head): softmax(q k^T/sqrt(32) with zero-attn) v
// Block = 256 threads (8 waves); wave w owns query rows w*16..w*16+15.
__global__ __launch_bounds__(256)
void k_attn(const half_t* __restrict__ q16, const half_t* __restrict__ k16,
            const half_t* __restrict__ vbuf, half_t* __restrict__ cp16) {
  int b = blockIdx.x / NHEADc, h = blockIdx.x % NHEADc;
  __shared__ half_t vT[HDHEAD * NPGc];         // v transposed: [dim][key]
  __shared__ half_t Psh[8 * 16 * NPGc];        // per-wave P tiles (f16)
  __shared__ float  dnl[8][16];

  int tid = threadIdx.x, wave = tid >> 5, lane = tid & 31;
  const half_t* vb = vbuf + (size_t)b * NPGc * HDIM + h * HDHEAD;
  for (int i = tid; i < HDHEAD * NPGc; i += 256) {
    int d = i >> 7, key = i & 127;
    vT[d * NPGc + key] = vb[(size_t)key * HDIM + d];
  }
  __syncthreads();

  int lrow = lane & 15;
  int qrow = wave * 16 + lrow;
  const half_t* qb = q16 + (size_t)b * NPGc * HDIM + h * HDHEAD;
  const half_t* kb = k16 + (size_t)b * NPGc * HDIM + h * HDHEAD;

  v16h aq = load_a_frag(qb + (size_t)qrow * HDIM, lane);

  const float scale = 0.17677669529663688f;  // 1/sqrt(32)
  int kkh = (lane < 16) ? 0 : 16;
  v8f sacc[8];
  v16h bkcur = *(const v16h*)(kb + (size_t)lrow * HDIM + kkh);
  sfor<0, 8>([&](auto J) {
    constexpr int j = decltype(J)::v;
    v16h bknext = bkcur;
    if constexpr (j < 7) {
      int key = (j + 1) * 16 + lrow;
      bknext = *(const v16h*)(kb + (size_t)key * HDIM + kkh);
    }
    v8f z = {0.f, 0.f, 0.f, 0.f, 0.f, 0.f, 0.f, 0.f};
    sacc[j] = __builtin_amdgcn_wmma_f32_16x16x32_f16(
        false, aq, false, bkcur, (short)0, z, (j < 7), false);
    bkcur = bknext;
  });

  int r0 = (lane < 16) ? 0 : 8;
  float rs[8];
#pragma unroll
  for (int r = 0; r < 8; r++) {
    float m = -3.0e38f;
#pragma unroll
    for (int j = 0; j < 8; j++) { sacc[j][r] *= scale; m = fmaxf(m, sacc[j][r]); }
#pragma unroll
    for (int s = 1; s < 16; s <<= 1) m = fmaxf(m, __shfl_xor(m, s, 32));
    m = fmaxf(m, 0.f);                               // zero-attn key score = 0
    float ssum = 0.f;
#pragma unroll
    for (int j = 0; j < 8; j++) { float e = __expf(sacc[j][r] - m); sacc[j][r] = e; ssum += e; }
#pragma unroll
    for (int s = 1; s < 16; s <<= 1) ssum += __shfl_xor(ssum, s, 32);
    rs[r] = ssum + __expf(-m);                       // zero-attn denominator term
  }
  if (lrow == 0) {
#pragma unroll
    for (int r = 0; r < 8; r++) dnl[wave][r0 + r] = rs[r];
  }
  // stage P (f16, row-major per wave)
#pragma unroll
  for (int j = 0; j < 8; j++) {
    int key = j * 16 + lrow;
#pragma unroll
    for (int r = 0; r < 8; r++)
      Psh[wave * 2048 + (r0 + r) * NPGc + key] = (half_t)sacc[j][r];
  }

  // out[16 x 32] = P[16 x 128] @ v[128 x 32] ; kc-outer so each P chunk is
  // loaded once and the dim-tile pair shares A (reuse hint on the first).
  int ko = (lane < 16) ? 0 : 16;
  v8f o0 = {0.f, 0.f, 0.f, 0.f, 0.f, 0.f, 0.f, 0.f};
  v8f o1 = o0;
  sfor<0, 4>([&](auto KC) {
    constexpr int kc = decltype(KC)::v;
    const half_t* pp = &Psh[wave * 2048 + lrow * NPGc + kc * 32];
    v16h ap = cat8(*(const v8h*)(pp + ko), *(const v8h*)(pp + 8 + ko));
    int kk = kc * 32 + kkh;
    v16h bv0 = *(const v16h*)(&vT[(size_t)lrow * NPGc + kk]);
    v16h bv1 = *(const v16h*)(&vT[(16 + lrow) * NPGc + kk]);
    o0 = __builtin_amdgcn_wmma_f32_16x16x32_f16(
        false, ap, false, bv0, (short)0, o0, true, false);   // reuse A next
    o1 = __builtin_amdgcn_wmma_f32_16x16x32_f16(
        false, ap, false, bv1, (short)0, o1, false, false);
  });
#pragma unroll
  for (int dt = 0; dt < 2; dt++) {
    v8f o = dt ? o1 : o0;
#pragma unroll
    for (int r = 0; r < 8; r++) {
      int M = r0 + r;
      float val = o[r] / dnl[wave][M];
      int pos = wave * 16 + M, dim = dt * 16 + lrow;
      cp16[(size_t)(b * NPGc + pos) * HDIM + h * HDHEAD + dim] = (half_t)val;
    }
  }
}

// ---------------------------------------------------------------------------
// score[n] = sigmoid(cross[n,:].linW + linb[li]) ; one wave per row
__global__ __launch_bounds__(256)
void k_score(const float* __restrict__ cross, const float* __restrict__ linW,
             const float* __restrict__ linb, int li, float* __restrict__ score) {
  int wave = threadIdx.x >> 5, lane = threadIdx.x & 31;
  int row = blockIdx.x * 8 + wave;
  float p = 0.f;
#pragma unroll
  for (int c = 0; c < 4; c++) {
    int col = c * 32 + lane;
    p += cross[(size_t)row * HDIM + col] * linW[col];
  }
#pragma unroll
  for (int s = 1; s < 32; s <<= 1) p += __shfl_xor(p, s, 32);
  if (lane == 0) {
    float z = p + linb[li];
    score[row] = 1.f / (1.f + __expf(-z));
  }
}

__global__ void k_mulscore(const float* __restrict__ h, const float* __restrict__ score,
                           float* __restrict__ y) {
  int i = blockIdx.x * blockDim.x + threadIdx.x;
  y[i] = h[i] * score[i >> 7];
}

// ---------------------------------------------------------------------------
// BatchNorm stats: sums[col] += x, sums[128+col] += x*x over all rows
__global__ __launch_bounds__(256)
void k_bn_stats(const float* __restrict__ y, float* __restrict__ sums) {
  __shared__ float s1[256], s2[256];
  int t = threadIdx.x, col = t & 127, rhalf = t >> 7;
  float ps = 0.f, pss = 0.f;
  for (int rr = rhalf; rr < 256; rr += 2) {
    float v = y[(size_t)(blockIdx.x * 256 + rr) * HDIM + col];
    ps += v; pss += v * v;
  }
  s1[t] = ps; s2[t] = pss;
  __syncthreads();
  if (t < 128) {
    atomicAdd(&sums[col], s1[t] + s1[t + 128]);
    atomicAdd(&sums[128 + col], s2[t] + s2[t + 128]);
  }
}

__global__ void k_bn_apply(const float* __restrict__ y, const float* __restrict__ sums,
                           const float* __restrict__ gam, const float* __restrict__ bet,
                           float* __restrict__ hout, half_t* __restrict__ hout16,
                           float* __restrict__ emb, int colblk) {
  int i = blockIdx.x * blockDim.x + threadIdx.x;
  int col = i & 127, row = i >> 7;
  float mu = sums[col] * (1.f / NNODE);
  float var = sums[128 + col] * (1.f / NNODE) - mu * mu;
  float v = gam[col] * (y[i] - mu) * rsqrtf(var + EPSc) + bet[col];
  hout[i] = v;
  hout16[i] = (half_t)v;
  emb[(size_t)row * (L1c * HDIM) + colblk * HDIM + col] = v;
}

// ---------------------------------------------------------------------------
// GATv2
__device__ __forceinline__ void atomicMaxF(float* addr, float val) {
  int* ai = (int*)addr;
  int cur = __float_as_int(*addr);
  while (__int_as_float(cur) < val) {
    int old = atomicCAS(ai, cur, __float_as_int(val));
    if (old == cur) break;
    cur = old;
  }
}

__global__ void k_gat_init(float* __restrict__ gout, float* __restrict__ mx,
                           float* __restrict__ dn) {
  int i = blockIdx.x * blockDim.x + threadIdx.x;
  gout[i] = 0.f;
  if (i < NNODE) { mx[i] = -3.0e38f; dn[i] = 0.f; }
}

// e = leakyrelu(xl[src]+xr[dst],0.2) . att ; one wave per edge
__global__ __launch_bounds__(256)
void k_gat_edge1(const int* __restrict__ ei, const float* __restrict__ xl,
                 const float* __restrict__ xr, const float* __restrict__ att,
                 float* __restrict__ eatt, float* __restrict__ mx) {
  int wave = threadIdx.x >> 5, lane = threadIdx.x & 31;
  int e = blockIdx.x * 8 + wave;
  int src, dst;
  if (e < NEDGE) { src = ei[e]; dst = ei[NEDGE + e]; }
  else           { src = dst = e - NEDGE; }
  float p = 0.f;
#pragma unroll
  for (int c = 0; c < 4; c++) {
    int col = c * 32 + lane;
    float a = xl[(size_t)src * HDIM + col] + xr[(size_t)dst * HDIM + col];
    a = a > 0.f ? a : 0.2f * a;
    p += a * att[col];
  }
#pragma unroll
  for (int s = 1; s < 32; s <<= 1) p += __shfl_xor(p, s, 32);
  if (lane == 0) { eatt[e] = p; atomicMaxF(&mx[dst], p); }
}

__global__ void k_gat_edge2(const int* __restrict__ ei, float* __restrict__ eatt,
                            const float* __restrict__ mx, float* __restrict__ dn) {
  int e = blockIdx.x * blockDim.x + threadIdx.x;
  int dst = (e < NEDGE) ? ei[NEDGE + e] : e - NEDGE;
  float ex = __expf(eatt[e] - mx[dst]);
  eatt[e] = ex;
  atomicAdd(&dn[dst], ex);
}

// scatter: gout[dst] += xl[src]*alpha ; one wave per edge, 4 cols/lane
__global__ __launch_bounds__(256)
void k_gat_edge3(const int* __restrict__ ei, const float* __restrict__ eatt,
                 const float* __restrict__ dn, const float* __restrict__ xl,
                 float* __restrict__ gout) {
  int wave = threadIdx.x >> 5, lane = threadIdx.x & 31;
  int e = blockIdx.x * 8 + wave;
  int src, dst;
  if (e < NEDGE) { src = ei[e]; dst = ei[NEDGE + e]; }
  else           { src = dst = e - NEDGE; }
  float alpha = eatt[e] / dn[dst];
#pragma unroll
  for (int c = 0; c < 4; c++) {
    int col = c * 32 + lane;
    atomicAdd(&gout[(size_t)dst * HDIM + col], xl[(size_t)src * HDIM + col] * alpha);
  }
}

__global__ void k_gat_fin(const float* __restrict__ gout, const float* __restrict__ bias,
                          float* __restrict__ h, half_t* __restrict__ h16) {
  int i = blockIdx.x * blockDim.x + threadIdx.x;
  float v = gout[i] + bias[i & 127];
  v = v > 0.f ? v : 0.f;
  h[i] = v;
  h16[i] = (half_t)v;
}

// ---------------------------------------------------------------------------
extern "C" void kernel_launch(void* const* d_in, const int* in_sizes, int n_in,
                              void* d_out, int out_size, void* d_ws, size_t ws_size,
                              hipStream_t stream) {
  (void)in_sizes; (void)n_in; (void)out_size; (void)ws_size;
  const float* x_s   = (const float*)d_in[0];
  const float* x_t   = (const float*)d_in[1];
  const int*   eis   = (const int*)d_in[2];
  const int*   eit   = (const int*)d_in[3];
  const float* pre_W = (const float*)d_in[6];
  const float* pre_b = (const float*)d_in[7];
  const float* Wq = (const float*)d_in[8],  *bq = (const float*)d_in[9];
  const float* Wk = (const float*)d_in[10], *bk = (const float*)d_in[11];
  const float* Wv = (const float*)d_in[12], *bv = (const float*)d_in[13];
  const float* Wo = (const float*)d_in[14], *bo = (const float*)d_in[15];
  const float* linW = (const float*)d_in[16], *linb = (const float*)d_in[17];
  const float* gs = (const float*)d_in[18], *bs = (const float*)d_in[19];
  const float* gt = (const float*)d_in[20], *bt = (const float*)d_in[21];
  const float* cWl = (const float*)d_in[22], *cbl = (const float*)d_in[23];
  const float* cWr = (const float*)d_in[24], *cbr = (const float*)d_in[25];
  const float* catt = (const float*)d_in[26], *cbias = (const float*)d_in[27];

  float* emb_s = (float*)d_out;
  float* emb_t = emb_s + (size_t)NNODE * L1c * HDIM;

  // --- workspace carve ---
  char* base = (char*)d_ws; size_t off = 0;
  auto alloc = [&](size_t bytes) -> void* {
    void* p = base + off; off += (bytes + 255) & ~(size_t)255; return p;
  };
  const size_t NH4 = (size_t)NNODE * HDIM * sizeof(float);
  const size_t NH2 = (size_t)NNODE * HDIM * sizeof(half_t);
  float*  hs   = (float*)alloc(NH4);
  float*  ht   = (float*)alloc(NH4);
  float*  xl   = (float*)alloc(NH4);
  float*  xr   = (float*)alloc(NH4);
  float*  gout = (float*)alloc(NH4);
  float*  tmpA = (float*)alloc(NH4);
  half_t* hs16 = (half_t*)alloc(NH2);
  half_t* ht16 = (half_t*)alloc(NH2);
  half_t* q16  = (half_t*)alloc(NH2);
  half_t* k16  = (half_t*)alloc(NH2);
  half_t* vb16 = (half_t*)alloc(NH2);
  half_t* cp16 = (half_t*)alloc(NH2);
  half_t* w16  = (half_t*)alloc((size_t)(1 + 4*4 + 2*3) * 16384 * sizeof(half_t));
  half_t* preW16 = w16;
  half_t* wq16 = preW16 + 16384;
  half_t* wk16 = wq16 + 4 * 16384;
  half_t* wv16 = wk16 + 4 * 16384;
  half_t* wo16 = wv16 + 4 * 16384;
  half_t* wl16 = wo16 + 4 * 16384;
  half_t* wr16 = wl16 + 3 * 16384;
  float* score = (float*)alloc((size_t)NNODE * sizeof(float));
  float* eatt  = (float*)alloc((size_t)ETOT * sizeof(float));
  float* mx    = (float*)alloc((size_t)NNODE * sizeof(float));
  float* dn    = (float*)alloc((size_t)NNODE * sizeof(float));
  float* sums  = (float*)alloc(256 * sizeof(float));

  const int NHE = NNODE * HDIM;   // 4M elements

  // --- weight / input conversions to f16 ---
  k_cast<<<NHE / 256, 256, 0, stream>>>(x_s, hs16, NHE);
  k_cast<<<NHE / 256, 256, 0, stream>>>(x_t, ht16, NHE);
  k_cast<<<64, 256, 0, stream>>>(pre_W, preW16, 16384);
  k_cast<<<256, 256, 0, stream>>>(Wq, wq16, 4 * 16384);
  k_cast<<<256, 256, 0, stream>>>(Wk, wk16, 4 * 16384);
  k_cast<<<256, 256, 0, stream>>>(Wv, wv16, 4 * 16384);
  k_cast<<<256, 256, 0, stream>>>(Wo, wo16, 4 * 16384);
  k_cast<<<192, 256, 0, stream>>>(cWl, wl16, 3 * 16384);
  k_cast<<<192, 256, 0, stream>>>(cWr, wr16, 3 * 16384);

  auto gemm = [&](const half_t* A, const half_t* W, const float* b,
                  float* o32, half_t* o16, int act) {
    k_gemm128<<<NNODE / 128, 256, 0, stream>>>(A, W, b, o32, o16, act);
  };

  // pre-linear (in-place f16 update is safe: each block rewrites only its own rows)
  gemm(hs16, preW16, pre_b, hs, hs16, 0);
  gemm(ht16, preW16, pre_b, ht, ht16, 0);

  auto mha = [&](int li) {
    gemm(hs16, wq16 + li * 16384, bq + li * HDIM, nullptr, q16, 0);
    gemm(ht16, wk16 + li * 16384, bk + li * HDIM, nullptr, k16, 0);
    gemm(ht16, wv16 + li * 16384, bv + li * HDIM, nullptr, vb16, 0);
    k_attn<<<NB * NHEADc, 256, 0, stream>>>(q16, k16, vb16, cp16);
    gemm(cp16, wo16 + li * 16384, bo + li * HDIM, tmpA, nullptr, 0);
    k_score<<<NNODE / 8, 256, 0, stream>>>(tmpA, linW + li * HDIM, linb, li, score);
    k_mulscore<<<NHE / 256, 256, 0, stream>>>(hs, score, tmpA);
    // BN on hs*score
    k_zero<<<1, 256, 0, stream>>>(sums, 256);
    k_bn_stats<<<128, 256, 0, stream>>>(tmpA, sums);
    k_bn_apply<<<NHE / 256, 256, 0, stream>>>(tmpA, sums, gs + li * HDIM, bs + li * HDIM,
                                              hs, hs16, emb_s, li);
    // BN on ht
    k_zero<<<1, 256, 0, stream>>>(sums, 256);
    k_bn_stats<<<128, 256, 0, stream>>>(ht, sums);
    k_bn_apply<<<NHE / 256, 256, 0, stream>>>(ht, sums, gt + li * HDIM, bt + li * HDIM,
                                              ht, ht16, emb_t, li);
  };

  auto gat = [&](int i, const int* ei, float* h, half_t* h16) {
    gemm(h16, wl16 + i * 16384, cbl + i * HDIM, xl, nullptr, 0);
    gemm(h16, wr16 + i * 16384, cbr + i * HDIM, xr, nullptr, 0);
    k_gat_init<<<NHE / 256, 256, 0, stream>>>(gout, mx, dn);
    k_gat_edge1<<<ETOT / 8, 256, 0, stream>>>(ei, xl, xr, catt + i * HDIM, eatt, mx);
    k_gat_edge2<<<ETOT / 256, 256, 0, stream>>>(ei, eatt, mx, dn);
    k_gat_edge3<<<ETOT / 8, 256, 0, stream>>>(ei, eatt, dn, xl, gout);
    k_gat_fin<<<NHE / 256, 256, 0, stream>>>(gout, cbias + i * HDIM, h, h16);
  };

  mha(0);
  for (int i = 0; i < NLYR; i++) {
    gat(i, eis, hs, hs16);
    gat(i, eit, ht, ht16);
    mha(i + 1);
  }
}
